// SemGCN_MDN_Graph_30683246362732
// MI455X (gfx1250) — matrix operand order repaired and checked by reading, not compiled, not run on hardware
//
#include <hip/hip_runtime.h>
#include <hip/hip_bf16.h>
#include <math.h>

typedef __attribute__((ext_vector_type(16))) _Float16 v16h;
typedef __attribute__((ext_vector_type(8)))  _Float16 v8h;
typedef __attribute__((ext_vector_type(8)))  float    v8f;

#define J16   16
#define HID   128
#define BT    4          // batch rows per block
#define RS    136        // LDS row stride in halves (128 + 8 pad -> conflict-free)
#define BUFH  (BT*J16*RS)
#define SCRS  24         // per-wave scratch row stride in halves (48B, 16B aligned, conflict-free)

// workspace layout (bytes)
#define WS_A   0         // 10 layers * 32 lanes * 32B  (A-operand-swizzled off-diag A, f16, K16..31=0)
#define WS_D   10240     // 10 * 16 f32 diagonal A[i][i]
#define WS_SS  10880     // 10 * 128 * {scale,shift} f32 (bias folded in)
#define WS_W   21504     // 640 entries * 1KB (B-operand-swizzled f16 weights)

__device__ __forceinline__ int went(int layer, int mat, int kc, int ft) {
  return (((layer * 2 + mat) * 4 + kc) * 8 + ft);
}

#define CAT16(lo,hi) __builtin_shufflevector((lo),(hi),0,1,2,3,4,5,6,7,8,9,10,11,12,13,14,15)
#define WMMA(a,b,c)  __builtin_amdgcn_wmma_f32_16x16x32_f16(false,(a),false,(b),(short)0,(c),false,false)

// ---------------------------------------------------------------------------
// Prep kernel: softmax A-matrices, BN fold, operand-layout weight swizzle.
// ---------------------------------------------------------------------------
__global__ __launch_bounds__(256) void semgcn_prep(
    const float* __restrict__ Win,  const float* __restrict__ e_in,
    const float* __restrict__ b_in, const float* __restrict__ bn_in,
    const float* __restrict__ Wmid, const float* __restrict__ e_mid,
    const float* __restrict__ b_mid,const float* __restrict__ bn_mid,
    const float* __restrict__ Wout, const float* __restrict__ e_out,
    const float* __restrict__ b_out, char* __restrict__ ws)
{
  __shared__ float A[10][16][16];
  const int t = threadIdx.x;
  // skeleton adjacency (EDGES + diagonal), bit j set in row i
  const unsigned short mb[16] = {0x0093,0x0007,0x000E,0x000C,0x0031,0x0070,0x0060,0x0181,
                                 0x4B80,0x0700,0x0600,0x1900,0x3800,0x3000,0xC100,0xC000};
  // 1) masked row softmax for 10 layers
  for (int u = t; u < 160; u += 256) {
    int layer = u / 16, i = u % 16;
    const float* e = (layer == 0) ? e_in : (layer <= 8 ? e_mid + (layer - 1) * 256 : e_out);
    unsigned m = mb[i];
    float mx = -3.0e38f;
    for (int j = 0; j < 16; j++) if ((m >> j) & 1) mx = fmaxf(mx, e[i * 16 + j]);
    float s = 0.f, w[16];
    for (int j = 0; j < 16; j++) {
      float v = ((m >> j) & 1) ? expf(e[i * 16 + j] - mx) : 0.f;
      w[j] = v; s += v;
    }
    float inv = 1.f / s;
    for (int j = 0; j < 16; j++) A[layer][i][j] = w[j] * inv;
  }
  __syncthreads();
  // 2) diagonal values
  for (int u = t; u < 160; u += 256) {
    int layer = u / 16, i = u % 16;
    ((float*)(ws + WS_D))[layer * 16 + i] = A[layer][i][i];
  }
  // 3) off-diagonal A in WMMA A-operand layout (16x32 f16, K=16..31 zero-padded)
  for (int u = t; u < 320; u += 256) {
    int layer = u / 32, lane = u % 32, row = lane % 16, hi = lane / 16;
    _Float16* dst = (_Float16*)(ws + WS_A + layer * 1024 + lane * 32);
    for (int v = 0; v < 8; v++)
      for (int h = 0; h < 2; h++) {
        int K = hi * 16 + 2 * v + h;
        float val = (K < 16 && K != row) ? A[layer][row][K] : 0.f;
        dst[v * 2 + h] = (_Float16)val;
      }
  }
  // 4) BN scale/shift with bias folded (layer 9: scale=1, shift=b_out)
  for (int u = t; u < 1280; u += 256) {
    int layer = u / 128, f = u % 128; float scale, shift;
    if (layer == 0) {
      float g = bn_in[f], be = bn_in[128 + f], mn = bn_in[256 + f], vr = bn_in[384 + f];
      scale = g * rsqrtf(vr + 1e-5f); shift = (b_in[f] - mn) * scale + be;
    } else if (layer <= 8) {
      int k = layer - 1; const float* bn = bn_mid + k * 4 * 128;
      float g = bn[f], be = bn[128 + f], mn = bn[256 + f], vr = bn[384 + f];
      scale = g * rsqrtf(vr + 1e-5f); shift = (b_mid[k * 128 + f] - mn) * scale + be;
    } else {
      scale = (f < 25) ? 1.f : 0.f; shift = (f < 25) ? b_out[f] : 0.f;
    }
    float* d = (float*)(ws + WS_SS);
    d[(layer * 128 + f) * 2] = scale; d[(layer * 128 + f) * 2 + 1] = shift;
  }
  // 5) weights in WMMA B-operand layout: entry(layer,mat,kc,ft), 32B per lane
  for (int u = t; u < 640 * 32; u += 256) {
    int lane = u % 32, ent = u / 32;
    int ft = ent % 8, kc = (ent / 8) % 4, mat = (ent / 32) % 2, layer = ent / 64;
    int col = ft * 16 + lane % 16, hi = lane / 16;
    _Float16* dst = (_Float16*)(ws + WS_W + (size_t)ent * 1024 + lane * 32);
    for (int v = 0; v < 8; v++)
      for (int h = 0; h < 2; h++) {
        int c = kc * 32 + ((v < 4) ? (hi * 8 + 2 * v + h) : (16 + hi * 8 + 2 * (v - 4) + h));   // B-operand K index
        float val = 0.f;
        if (layer == 0)      { if (c < 2) val = Win[(mat * 2 + c) * 128 + col]; }
        else if (layer <= 8) { val = Wmid[(((size_t)(layer - 1) * 2 + mat) * 128 + c) * 128 + col]; }
        else                 { if (col < 25) val = Wout[((size_t)mat * 128 + c) * 25 + col]; }
        dst[v * 2 + h] = (_Float16)val;
      }
  }
}

// ---------------------------------------------------------------------------
// One SemGConv layer (GEMM x2 via WMMA + aggregation via WMMA + BN/ReLU/res)
// ---------------------------------------------------------------------------
template <int KCN>
__device__ __forceinline__ void conv_layer(
    int layer, const _Float16* src, _Float16* dst,
    const _Float16* res, const char* __restrict__ ws,
    _Float16* scrw, int wave, int lane)
{
  const int hi = lane >> 4, ln = lane & 15, ft = wave;
  const int f = ft * 16 + ln;
  v16h W0B[KCN], W1B[KCN];
#pragma unroll
  for (int kc = 0; kc < KCN; kc++) {
    const v8h* p0 = (const v8h*)(ws + WS_W + (size_t)went(layer, 0, kc, ft) * 1024 + lane * 32);
    const v8h* p1 = (const v8h*)(ws + WS_W + (size_t)went(layer, 1, kc, ft) * 1024 + lane * 32);
    W0B[kc] = CAT16(p0[0], p0[1]);
    W1B[kc] = CAT16(p1[0], p1[1]);
  }
  const v8h* pa = (const v8h*)(ws + WS_A + layer * 1024 + lane * 32);
  const v16h Aop = CAT16(pa[0], pa[1]);
  float dv[8];
  const float* wd = (const float*)(ws + WS_D) + layer * 16;
#pragma unroll
  for (int r = 0; r < 8; r++) dv[r] = wd[hi * 8 + r];
  const float* ssp = (const float*)(ws + WS_SS) + ((size_t)layer * 128 + f) * 2;
  const float scl = ssp[0], shf = ssp[1];

  for (int b = 0; b < BT; b++) {
    v8f acc0 = {}, acc1 = {};
    const _Float16* srow = src + (b * 16 + ln) * RS;   // A-operand row M = lane%16
#pragma unroll
    for (int kc = 0; kc < KCN; kc++) {
      const _Float16* ap = srow + kc * 32 + hi * 8;
      v8h alo = *(const v8h*)ap;
      v8h ahi = *(const v8h*)(ap + 16);
      v16h a = CAT16(alo, ahi);
      acc0 = WMMA(a, W0B[kc], acc0);   // g0 = h @ W0
      acc1 = WMMA(a, W1B[kc], acc1);   // g1 = h @ W1
    }
    // bounce g1 through LDS: C-layout -> transposed (f_local, j) -> B-operand
    v8h pk;
#pragma unroll
    for (int r = 0; r < 8; r++) pk[r] = (_Float16)acc1[r];
    *(v8h*)(scrw + ln * SCRS + hi * 8) = pk;
    v16h g1b = {};
    if (lane < 16) {
      v8h lo = *(const v8h*)(scrw + lane * SCRS);
      v8h h2 = *(const v8h*)(scrw + lane * SCRS + 8);
      g1b = CAT16(lo, h2);              // lanes 16..31 stay zero (K padding)
    }
    v8f zc = {};
    v8f agg = WMMA(Aop, g1b, zc);       // sum_j A'[i,j] * g1[j,f]
    // epilogue: diag term + BN(+bias) + ReLU (+ residual), store f16
#pragma unroll
    for (int r = 0; r < 8; r++) {
      float v = agg[r] + dv[r] * acc0[r];
      v = fmaxf(v * scl + shf, 0.f);
      int off = (b * 16 + (hi * 8 + r)) * RS + f;
      if (res) v += (float)res[off];
      dst[off] = (_Float16)v;
    }
  }
}

// ---------------------------------------------------------------------------
// Fused 10-layer SemGCN + MDN heads. 4096 blocks x 256 threads (8 wave32).
// ---------------------------------------------------------------------------
__global__ __launch_bounds__(256) void semgcn_main(
    const float* __restrict__ x, const char* __restrict__ ws, float* __restrict__ out)
{
  extern __shared__ __align__(16) _Float16 smem[];
  // NOTE: no brace-initialized array of LDS pointers (ld.lld addrspacecast issue);
  // compute buffer bases on demand instead.
#define BUF(i) (smem + (size_t)(i) * BUFH)
  _Float16* scrbase = smem + 3 * BUFH;
  const int tid = threadIdx.x, lane = tid & 31, wave = tid >> 5;
  const int hi = lane >> 4, ln = lane & 15;
  _Float16* scrw = scrbase + wave * (16 * SCRS);
  const int blockb = blockIdx.x;

  // zero buf0 (input layer K-padding), then load x as f16
  for (int i = tid; i < BUFH; i += 256) BUF(0)[i] = (_Float16)0.f;
  __syncthreads();
  if (tid < BT * 16 * 2) {
    int b = tid >> 5, j = (tid >> 1) & 15, c = tid & 1;
    BUF(0)[(b * 16 + j) * RS + c] =
        (_Float16)x[(((size_t)blockb * BT + b) * 16 + j) * 2 + c];
  }
  __syncthreads();

  // input layer (K=2 padded to 32): buf0 -> buf1
  conv_layer<1>(0, BUF(0), BUF(1), nullptr, ws, scrw, wave, lane);
  __syncthreads();

  int cur = 1;
  for (int blk = 0; blk < 4; blk++) {
    int resb = cur;
    int t1 = (cur + 1) % 3;
    int t2 = 3 - cur - t1;
    conv_layer<4>(1 + 2 * blk, BUF(cur), BUF(t1), nullptr, ws, scrw, wave, lane);
    __syncthreads();
    conv_layer<4>(2 + 2 * blk, BUF(t1), BUF(t2), BUF(resb), ws, scrw, wave, lane);
    __syncthreads();
    cur = t2;
  }

  // output layer: N=25 padded to 32 -> 2 f-tiles; wave w: ft=w&1, b=w>>1
  {
    const int layer = 9, ft = wave & 1, b = wave >> 1;
    const int f = ft * 16 + ln;
    v16h W0B[4], W1B[4];
#pragma unroll
    for (int kc = 0; kc < 4; kc++) {
      const v8h* p0 = (const v8h*)(ws + WS_W + (size_t)went(layer, 0, kc, ft) * 1024 + lane * 32);
      const v8h* p1 = (const v8h*)(ws + WS_W + (size_t)went(layer, 1, kc, ft) * 1024 + lane * 32);
      W0B[kc] = CAT16(p0[0], p0[1]);
      W1B[kc] = CAT16(p1[0], p1[1]);
    }
    const v8h* pa = (const v8h*)(ws + WS_A + layer * 1024 + lane * 32);
    v16h Aop = CAT16(pa[0], pa[1]);
    float dv[8];
    const float* wd = (const float*)(ws + WS_D) + layer * 16;
#pragma unroll
    for (int r = 0; r < 8; r++) dv[r] = wd[hi * 8 + r];
    const float* ssp = (const float*)(ws + WS_SS) + ((size_t)layer * 128 + f) * 2;
    float scl = ssp[0], shf = ssp[1];

    v8f acc0 = {}, acc1 = {};
    const _Float16* srow = BUF(cur) + (b * 16 + ln) * RS;
#pragma unroll
    for (int kc = 0; kc < 4; kc++) {
      const _Float16* ap = srow + kc * 32 + hi * 8;
      v8h alo = *(const v8h*)ap;
      v8h ahi = *(const v8h*)(ap + 16);
      v16h a = CAT16(alo, ahi);
      acc0 = WMMA(a, W0B[kc], acc0);
      acc1 = WMMA(a, W1B[kc], acc1);
    }
    v8h pk;
#pragma unroll
    for (int r = 0; r < 8; r++) pk[r] = (_Float16)acc1[r];
    *(v8h*)(scrw + ln * SCRS + hi * 8) = pk;
    v16h g1b = {};
    if (lane < 16) {
      v8h lo = *(const v8h*)(scrw + lane * SCRS);
      v8h h2 = *(const v8h*)(scrw + lane * SCRS + 8);
      g1b = CAT16(lo, h2);
    }
    v8f zc = {};
    v8f agg = WMMA(Aop, g1b, zc);

    const size_t MU = 3932160u;   // B*16*15
    const size_t SG = 1310720u;   // B*16*5
#pragma unroll
    for (int r = 0; r < 8; r++) {
      float v = agg[r] + dv[r] * acc0[r];
      v = v * scl + shf;
      int n = f, i = hi * 8 + r;
      size_t row = ((size_t)blockb * BT + b) * 16 + i;
      if (n < 15)      out[row * 15 + n] = tanhf(v);
      else if (n < 20) out[MU + row * 5 + (n - 15)] =
                           fmaxf((v > 0.f) ? (v + 1.f) : expf(v), 1e-10f);
      else if (n < 25) out[MU + SG + row * 5 + (n - 20)] = v;
    }
  }
#undef BUF
}

extern "C" void kernel_launch(void* const* d_in, const int* in_sizes, int n_in,
                              void* d_out, int out_size, void* d_ws, size_t ws_size,
                              hipStream_t stream) {
  const float* x      = (const float*)d_in[0];
  const float* Win    = (const float*)d_in[1];
  const float* e_in   = (const float*)d_in[2];
  const float* b_in   = (const float*)d_in[3];
  const float* bn_in  = (const float*)d_in[4];
  const float* Wmid   = (const float*)d_in[5];
  const float* e_mid  = (const float*)d_in[6];
  const float* b_mid  = (const float*)d_in[7];
  const float* bn_mid = (const float*)d_in[8];
  const float* Wout   = (const float*)d_in[9];
  const float* e_out  = (const float*)d_in[10];
  const float* b_out  = (const float*)d_in[11];
  char*  ws  = (char*)d_ws;
  float* out = (float*)d_out;

  semgcn_prep<<<1, 256, 0, stream>>>(Win, e_in, b_in, bn_in, Wmid, e_mid,
                                     b_mid, bn_mid, Wout, e_out, b_out, ws);
  size_t sh = (size_t)(3 * BUFH + 8 * 16 * SCRS) * sizeof(_Float16);  // 58,368 B
  semgcn_main<<<16384 / BT, 256, sh, stream>>>(x, ws, out);
}
